// LAINRDecoder_53085795779238
// MI455X (gfx1250) — compile-verified, matrix operand order checked by reading, not run on hardware
//
#include <hip/hip_runtime.h>
#include <hip/hip_bf16.h>
#include <stdint.h>

// Problem constants (match reference)
#define BB     4
#define LL     16384
#define MM     256
#define HIDD   512
#define FDIMM  128
#define ODIMM  3
#define NHEAD  2
#define DHH    64
#define PI_F   3.14159265358979f

typedef __bf16 bf16_t;
typedef __attribute__((ext_vector_type(16))) __bf16 bf16x16;
typedef __attribute__((ext_vector_type(8)))  float  f32x8;
typedef __attribute__((ext_vector_type(4)))  unsigned int u32x4;
typedef __attribute__((ext_vector_type(8)))  int i32x8;
typedef __attribute__((ext_vector_type(4)))  int i32x4;

union FragU { bf16x16 v; uint4 q[2]; };

__device__ __forceinline__ bf16_t f2bf(float f) { return (bf16_t)f; }

// A-fragment (or row-major B with "column index = lane") loader.
// CDNA5 16-bit A 16x32 layout: lanes 0-15: m=lane, K in {kb..kb+7, kb+16..kb+23}
//                               lanes 16-31: m=lane-16, K in {kb+8..kb+15, kb+24..kb+31}
__device__ __forceinline__ bf16x16 load_frag(const bf16_t* base, int stride, int kbase) {
  const int lane = threadIdx.x & 31;
  const int m    = lane & 15;
  const int kh8  = (lane >> 4) << 3;
  const bf16_t* p = base + (size_t)m * stride + kbase + kh8;
  FragU f;
  f.q[0] = *(const uint4*)(p);
  f.q[1] = *(const uint4*)(p + 16);
  return f.v;
}

// Pre-swizzled (fragment-packed) weight B loader: 16 contiguous bf16 per lane.
__device__ __forceinline__ bf16x16 load_fragB(const bf16_t* wp, int kt, int nt, int NT) {
  const int lane = threadIdx.x & 31;
  const bf16_t* p = wp + ((size_t)(kt * NT + nt) * 32 + lane) * 16;
  FragU f;
  f.q[0] = *(const uint4*)(p);
  f.q[1] = *(const uint4*)(p + 8);
  return f.v;
}

__device__ __forceinline__ f32x8 wmma_bf(bf16x16 a, bf16x16 b, f32x8 c) {
  return __builtin_amdgcn_wmma_f32_16x16x32_bf16(false, a, false, b, (short)0, c, false, false);
}

#define ZERO8 {0.f,0.f,0.f,0.f,0.f,0.f,0.f,0.f}

// ---------------------------------------------------------------------------
// Tensor Data Mover: 1-D contiguous copy global -> LDS.
// D# per CDNA5 ISA: group0 = {count/lds_addr/global_addr/type=2},
// group1 = {data_size=8B, tensor_dim0 = tile_dim0 = nbytes/8, dim1=1, stride}.
// Issued by one wave; completion via TENSORcnt.
// ---------------------------------------------------------------------------
#if defined(__has_builtin)
#if __has_builtin(__builtin_amdgcn_tensor_load_to_lds)
#define HAVE_TDM 1
#endif
#endif

__device__ __forceinline__ void tdm_load_1d(unsigned lds_off, const void* gptr,
                                            unsigned nbytes) {
#ifdef HAVE_TDM
  const unsigned n8 = nbytes >> 3;  // 8-byte elements
  const unsigned long long ga = (unsigned long long)(uintptr_t)gptr;
  u32x4 g0 = {1u,                                    // count=1, user mode
              lds_off,                               // lds_addr (bytes)
              (unsigned)(ga & 0xFFFFFFFFu),          // global_addr[31:0]
              (unsigned)((ga >> 32) & 0x01FFFFFFu) | (2u << 30)};  // ga[56:32], type=2
  i32x8 g1;
  g1[0] = (int)(3u << 16);                           // wg_mask=0, data_size=3 (8B)
  g1[1] = (int)((n8 & 0xFFFFu) << 16);               // tensor_dim0[15:0] @ bits 63:48
  g1[2] = (int)(((n8 >> 16) & 0xFFFFu) | (1u << 16));// tensor_dim0[31:16], tensor_dim1=1
  g1[3] = (int)((n8 & 0xFFFFu) << 16);               // tile_dim0 @ bits 127:112
  g1[4] = 1;                                         // tile_dim1=1, tile_dim2=0
  g1[5] = (int)n8;                                   // tensor_dim0_stride[31:0]
  g1[6] = 0;                                         // stride hi / dim1_stride lo
  g1[7] = 0;
  i32x4 z4 = {0, 0, 0, 0};
#if __has_include(<hip/amd_detail/amd_gfx1250_TDM.h>)
  i32x8 z8 = {0, 0, 0, 0, 0, 0, 0, 0};
  __builtin_amdgcn_tensor_load_to_lds(g0, g1, z4, z4, z8, 0);
#else
  __builtin_amdgcn_tensor_load_to_lds(g0, g1, z4, z4, 0);
#endif
  __builtin_amdgcn_s_wait_tensorcnt(0);
#else
  // Fallback: no-op; the caller copies cooperatively with all threads of the block.
  bf16_t* dst = (bf16_t*)(uintptr_t)0;  // unused; fallback handled by caller
  (void)lds_off; (void)gptr; (void)nbytes; (void)dst;
#endif
}

// ---------------------------------------------------------------------------
// Pack f32 [K][N] weight into WMMA-B fragment layout, bf16.
// ---------------------------------------------------------------------------
__global__ __launch_bounds__(256) void pack_weight_kernel(const float* __restrict__ src,
                                                          bf16_t* __restrict__ dst,
                                                          int K, int N) {
  int tid = blockIdx.x * blockDim.x + threadIdx.x;
  int NT = N >> 4, KT = K >> 5;
  int total = KT * NT * 32;
  if (tid >= total) return;
  int lane = tid & 31;
  int blk  = tid >> 5;
  int nt = blk % NT, kt = blk / NT;
  int n   = nt * 16 + (lane & 15);
  int kh8 = (lane >> 4) << 3;
  bf16_t* out = dst + (size_t)blk * 512 + lane * 16;
#pragma unroll
  for (int i = 0; i < 8; ++i)
    out[i] = f2bf(src[(size_t)(kt * 32 + kh8 + i) * N + n]);
#pragma unroll
  for (int i = 0; i < 8; ++i)
    out[8 + i] = f2bf(src[(size_t)(kt * 32 + 16 + kh8 + i) * N + n]);
}

__global__ __launch_bounds__(256) void f32_to_bf16_kernel(const float* __restrict__ src,
                                                          bf16_t* __restrict__ dst, int n) {
  int i = blockIdx.x * blockDim.x + threadIdx.x;
  if (i < n) dst[i] = f2bf(src[i]);
}

// t[l] (relative position scalar for ALiBi-like bias)
__global__ __launch_bounds__(256) void tpos_kernel(const float* __restrict__ x,
                                                   float* __restrict__ tpos,
                                                   const int* Dd, const int* Hh,
                                                   const int* Ww, const int* Tt) {
  int l = blockIdx.x * blockDim.x + threadIdx.x;
  if (l >= LL) return;
  int D = *Dd, H = *Hh, W = *Ww, T = *Tt;
  const float* g = x + (size_t)l * 4;
  int z  = (int)(g[0] * D);
  int y  = (int)(g[1] * H);
  int xi = (int)(g[2] * W);
  int t  = (int)(g[3] * T);
  int idx = ((t * D + z) * H + y) * W + xi;
  tpos[l] = (float)idx / (float)(D * H * W * T);
}

// init d_out with sum_k b_out[k][o]
__global__ __launch_bounds__(256) void out_init_kernel(float* __restrict__ out,
                                                       const float* __restrict__ b_out) {
  int i = blockIdx.x * blockDim.x + threadIdx.x;
  if (i >= BB * LL * ODIMM) return;
  int o = i % ODIMM;
  out[i] = b_out[o] + b_out[ODIMM + o] + b_out[2 * ODIMM + o];
}

// ---------------------------------------------------------------------------
// Feature kernel: gamma -> x_q -> q, and 3 band features h_l. 32 rows/block.
// ---------------------------------------------------------------------------
__device__ __forceinline__ void fill_gamma(bf16_t* sg, const float* xg, int row0, float sigma) {
  float d = (log10f(sigma) - 1.0f) * (1.0f / 15.0f);
  for (int i = threadIdx.x; i < 32 * 128; i += blockDim.x) {
    int row = i >> 7, col = i & 127;
    int c = col >> 5, rem = col & 31;
    int f = rem & 15, isc = rem >> 4;
    float omega = __powf(10.f, 1.0f + (float)f * d);
    float arg = PI_F * xg[(size_t)(row0 + row) * 4 + c] * omega;
    sg[i] = f2bf(isc ? __cosf(arg) : __sinf(arg));
  }
}

__device__ __forceinline__ void gemm128_512(const bf16_t* sg, const bf16_t* Wp,
                                            const float* __restrict__ bias, bool relu,
                                            bf16_t* dst /* row_local*512 indexed */) {
  const int w = threadIdx.x >> 5, lane = threadIdx.x & 31;
  const int rt = w & 1, ct0 = (w >> 1) * 8;
  const int kh8 = (lane >> 4) << 3;
  f32x8 acc[8] = {ZERO8, ZERO8, ZERO8, ZERO8, ZERO8, ZERO8, ZERO8, ZERO8};
#pragma unroll
  for (int ks = 0; ks < 4; ++ks) {
    bf16x16 a = load_frag(sg + rt * 16 * 128, 128, ks * 32);
#pragma unroll
    for (int t = 0; t < 8; ++t)
      acc[t] = wmma_bf(a, load_fragB(Wp, ks, ct0 + t, 32), acc[t]);
  }
#pragma unroll
  for (int t = 0; t < 8; ++t) {
    int col = (ct0 + t) * 16 + (lane & 15);
    float bv = bias[col];
#pragma unroll
    for (int r = 0; r < 8; ++r) {
      int row = rt * 16 + r + kh8;
      float v = acc[t][r] + bv;
      if (relu) v = fmaxf(v, 0.f);
      dst[(size_t)row * 512 + col] = f2bf(v);
    }
  }
}

__global__ __launch_bounds__(256) void feature_kernel(
    const float* __restrict__ x, const bf16_t* __restrict__ wq_p,
    const float* __restrict__ b_query, const bf16_t* __restrict__ wband_p,
    const float* __restrict__ b_band, const bf16_t* __restrict__ wtoq_p,
    bf16_t* __restrict__ qb, bf16_t* __restrict__ hband) {
  __shared__ bf16_t s_gamma[32 * 128];
  __shared__ bf16_t s_xq[32 * 512];
  const int row0 = blockIdx.x * 32;
  const int w = threadIdx.x >> 5, lane = threadIdx.x & 31;
  const int kh8 = (lane >> 4) << 3;

  // x_q = relu(gamma(sigma_q=128) @ w_query + b_query)
  fill_gamma(s_gamma, x, row0, 128.0f);
  __syncthreads();
  gemm128_512(s_gamma, wq_p, b_query, true, s_xq);
  __syncthreads();

  // q = x_q @ w_to_q  (no bias) -> qb [L][128]
  {
    const int rt = w & 1, ct0 = (w >> 1) * 2;
    f32x8 acc[2] = {ZERO8, ZERO8};
#pragma unroll
    for (int ks = 0; ks < 16; ++ks) {
      bf16x16 a = load_frag(s_xq + rt * 16 * 512, 512, ks * 32);
#pragma unroll
      for (int t = 0; t < 2; ++t)
        acc[t] = wmma_bf(a, load_fragB(wtoq_p, ks, ct0 + t, 8), acc[t]);
    }
#pragma unroll
    for (int t = 0; t < 2; ++t) {
      int col = (ct0 + t) * 16 + (lane & 15);
#pragma unroll
      for (int r = 0; r < 8; ++r) {
        int row = row0 + rt * 16 + r + kh8;
        qb[(size_t)row * 128 + col] = f2bf(acc[t][r]);
      }
    }
  }

  // band features h_l[k] = relu(gamma(sigma_k) @ w_band[k] + b_band[k])
  const float sig[3] = {128.f, 32.f, 8.f};
#pragma unroll
  for (int k = 0; k < 3; ++k) {
    __syncthreads();
    fill_gamma(s_gamma, x, row0, sig[k]);
    __syncthreads();
    gemm128_512(s_gamma, wband_p + (size_t)k * 128 * 512, b_band + k * 512, true,
                hband + ((size_t)k * LL + row0) * 512);
  }
}

// ---------------------------------------------------------------------------
// kv = tokens @ w_to_kv ; split to k [B][H][M][64] and vT [B][H][64][M], bf16.
// ---------------------------------------------------------------------------
__global__ __launch_bounds__(256) void kv_kernel(const bf16_t* __restrict__ tok_bf,
                                                 const bf16_t* __restrict__ wtokv_p,
                                                 bf16_t* __restrict__ kb,
                                                 bf16_t* __restrict__ vT) {
  const int w = threadIdx.x >> 5, lane = threadIdx.x & 31;
  const int rt = w & 1, ct0 = (w >> 1) * 4;  // NT=16
  const int kh8 = (lane >> 4) << 3;
  const int row0 = blockIdx.x * 32;
  const bf16_t* A = tok_bf + (size_t)(row0 + rt * 16) * 512;
  f32x8 acc[4] = {ZERO8, ZERO8, ZERO8, ZERO8};
#pragma unroll
  for (int ks = 0; ks < 16; ++ks) {
    bf16x16 a = load_frag(A, 512, ks * 32);
#pragma unroll
    for (int t = 0; t < 4; ++t)
      acc[t] = wmma_bf(a, load_fragB(wtokv_p, ks, ct0 + t, 16), acc[t]);
  }
#pragma unroll
  for (int t = 0; t < 4; ++t) {
    int c = (ct0 + t) * 16 + (lane & 15);
#pragma unroll
    for (int r = 0; r < 8; ++r) {
      int grow = row0 + rt * 16 + r + kh8;
      int b = grow >> 8, m = grow & 255;
      float v = acc[t][r];
      if (c < 128) {
        int h = c >> 6, d = c & 63;
        kb[(((size_t)(b * 2 + h) * MM + m) * DHH) + d] = f2bf(v);
      } else {
        int c2 = c - 128, h = c2 >> 6, d = c2 & 63;
        vT[(((size_t)(b * 2 + h) * DHH + d) * MM) + m] = f2bf(v);
      }
    }
  }
}

// ---------------------------------------------------------------------------
// Attention: per block (64 rows of L, one (b,h)). 4 waves of 16 rows each.
// K tile (32KB) staged into LDS via Tensor Data Mover; attn staged in LDS.
// ---------------------------------------------------------------------------
__global__ __launch_bounds__(128) void attn_kernel(const bf16_t* __restrict__ qb,
                                                   const bf16_t* __restrict__ kb,
                                                   const bf16_t* __restrict__ vT,
                                                   const float* __restrict__ tpos,
                                                   bf16_t* __restrict__ attn_out) {
  __shared__ bf16_t s_attn[4][16][256];  // 32 KB
  __shared__ bf16_t s_k[MM * DHH];       // 32 KB, K tile for this (b,h)
  const int bh = blockIdx.y, b = bh >> 1, h = bh & 1;
  const int w = threadIdx.x >> 5, lane = threadIdx.x & 31;
  const int kh8 = (lane >> 4) << 3;
  const int row0 = blockIdx.x * 64 + w * 16;

  // Stage K via TDM (wave 0 issues; descriptor values are wave-uniform -> SGPRs)
#ifdef HAVE_TDM
  if (w == 0) {
    tdm_load_1d((unsigned)(uintptr_t)&s_k[0], kb + (size_t)bh * MM * DHH,
                MM * DHH * sizeof(bf16_t));
  }
#else
  for (int i = threadIdx.x * 8; i < MM * DHH; i += blockDim.x * 8)
    *(uint4*)&s_k[i] = *(const uint4*)&kb[(size_t)bh * MM * DHH + i];
#endif
  __syncthreads();

  f32x8 acc[16] = {ZERO8, ZERO8, ZERO8, ZERO8, ZERO8, ZERO8, ZERO8, ZERO8,
                   ZERO8, ZERO8, ZERO8, ZERO8, ZERO8, ZERO8, ZERO8, ZERO8};
  const bf16_t* Aq = qb + (size_t)row0 * 128 + h * 64;
#pragma unroll
  for (int ks = 0; ks < 2; ++ks) {
    bf16x16 a = load_frag(Aq, 128, ks * 32);
#pragma unroll
    for (int nt = 0; nt < 16; ++nt)
      acc[nt] = wmma_bf(a, load_frag(&s_k[nt * 16 * DHH], DHH, ks * 32), acc[nt]);
  }

  // bias + scale
  float tv[8];
#pragma unroll
  for (int r = 0; r < 8; ++r) tv[r] = tpos[row0 + r + kh8];
#pragma unroll
  for (int nt = 0; nt < 16; ++nt) {
    float tp = ((float)(nt * 16 + (lane & 15)) + 0.5f) * (1.0f / 256.0f);
#pragma unroll
    for (int r = 0; r < 8; ++r) {
      float d = tv[r] - tp;
      acc[nt][r] = acc[nt][r] * 0.125f - 10.0f * d * d;
    }
  }

  // softmax across 256 cols (16 tiles x 16 lanes of half-group)
  float mx[8], sm[8];
#pragma unroll
  for (int r = 0; r < 8; ++r) mx[r] = -1e30f;
#pragma unroll
  for (int nt = 0; nt < 16; ++nt)
#pragma unroll
    for (int r = 0; r < 8; ++r) mx[r] = fmaxf(mx[r], acc[nt][r]);
#pragma unroll
  for (int off = 1; off < 16; off <<= 1)
#pragma unroll
    for (int r = 0; r < 8; ++r) mx[r] = fmaxf(mx[r], __shfl_xor(mx[r], off, 32));
#pragma unroll
  for (int r = 0; r < 8; ++r) sm[r] = 0.f;
#pragma unroll
  for (int nt = 0; nt < 16; ++nt)
#pragma unroll
    for (int r = 0; r < 8; ++r) {
      float e = __expf(acc[nt][r] - mx[r]);
      acc[nt][r] = e;
      sm[r] += e;
    }
#pragma unroll
  for (int off = 1; off < 16; off <<= 1)
#pragma unroll
    for (int r = 0; r < 8; ++r) sm[r] += __shfl_xor(sm[r], off, 32);
#pragma unroll
  for (int r = 0; r < 8; ++r) sm[r] = 1.0f / sm[r];

  // stage attn (bf16) into LDS, per wave
#pragma unroll
  for (int nt = 0; nt < 16; ++nt)
#pragma unroll
    for (int r = 0; r < 8; ++r)
      s_attn[w][r + kh8][nt * 16 + (lane & 15)] = f2bf(acc[nt][r] * sm[r]);
  __syncthreads();

  // out = attn @ v
  f32x8 oacc[4] = {ZERO8, ZERO8, ZERO8, ZERO8};
  const bf16_t* Vb = vT + (size_t)bh * DHH * MM;
#pragma unroll
  for (int ks = 0; ks < 8; ++ks) {
    bf16x16 a = load_frag(&s_attn[w][0][0], 256, ks * 32);
#pragma unroll
    for (int nt = 0; nt < 4; ++nt)
      oacc[nt] = wmma_bf(a, load_frag(Vb + nt * 16 * MM, MM, ks * 32), oacc[nt]);
  }
#pragma unroll
  for (int nt = 0; nt < 4; ++nt)
#pragma unroll
    for (int r = 0; r < 8; ++r) {
      int row = row0 + r + kh8;
      attn_out[((size_t)b * LL + row) * 128 + h * 64 + nt * 16 + (lane & 15)] =
          f2bf(oacc[nt][r]);
    }
}

// ---------------------------------------------------------------------------
// Fused chain: modv + 3 levels of 512x512 GEMMs with hv residual in LDS.
// ---------------------------------------------------------------------------
__device__ __forceinline__ void gemm512(const bf16_t* Alds, const bf16_t* Wp,
                                        int rt, int ct0, f32x8 acc[8]) {
#pragma unroll
  for (int t = 0; t < 8; ++t) acc[t] = (f32x8)ZERO8;
#pragma unroll
  for (int ks = 0; ks < 16; ++ks) {
    bf16x16 a = load_frag(Alds + rt * 16 * 512, 512, ks * 32);
#pragma unroll
    for (int t = 0; t < 8; ++t)
      acc[t] = wmma_bf(a, load_fragB(Wp, ks, ct0 + t, 32), acc[t]);
  }
}

__global__ __launch_bounds__(256) void chain_kernel(
    const bf16_t* __restrict__ attn_out, const bf16_t* __restrict__ wtoout_p,
    const float* __restrict__ b_to_out, const bf16_t* __restrict__ hband,
    const bf16_t* __restrict__ wmod_p, const float* __restrict__ b_mod,
    const bf16_t* __restrict__ whv_p, const float* __restrict__ b_hv,
    const float* __restrict__ w_out, float* __restrict__ outp) {
  __shared__ bf16_t s_modv[32 * 512];
  __shared__ bf16_t s_hv[32 * 512];
  const int w = threadIdx.x >> 5, lane = threadIdx.x & 31;
  const int rt = w & 1, ct0 = (w >> 1) * 8;
  const int kh8 = (lane >> 4) << 3;
  const int gRow0 = blockIdx.x * 32;

  f32x8 acc[8];
  float tot[8][3];
#pragma unroll
  for (int r = 0; r < 8; ++r)
#pragma unroll
    for (int o = 0; o < 3; ++o) tot[r][o] = 0.f;

  // warm L2/WGP$ for first weight slab
  __builtin_prefetch((const void*)wmod_p, 0, 1);

  // ---- modv = attn_out @ w_to_out + b_to_out (K=128) ----
  {
#pragma unroll
    for (int t = 0; t < 8; ++t) acc[t] = (f32x8)ZERO8;
    const bf16_t* A = attn_out + (size_t)(gRow0 + rt * 16) * 128;
#pragma unroll
    for (int ks = 0; ks < 4; ++ks) {
      bf16x16 a = load_frag(A, 128, ks * 32);
#pragma unroll
      for (int t = 0; t < 8; ++t)
        acc[t] = wmma_bf(a, load_fragB(wtoout_p, ks, ct0 + t, 32), acc[t]);
    }
#pragma unroll
    for (int t = 0; t < 8; ++t) {
      int col = (ct0 + t) * 16 + (lane & 15);
      float bv = b_to_out[col];
#pragma unroll
      for (int r = 0; r < 8; ++r) {
        int row = rt * 16 + r + kh8;
        s_modv[(size_t)row * 512 + col] = f2bf(acc[t][r] + bv);
      }
    }
  }
  __syncthreads();

#pragma unroll 1
  for (int k = 0; k < 3; ++k) {
    // m = relu(hband_k + modv @ w_mod[k] + b_mod[k])
    if (k > 0)
      __builtin_prefetch((const void*)(whv_p + (size_t)(k - 1) * 512 * 512), 0, 1);
    gemm512(s_modv, wmod_p + (size_t)k * 512 * 512, rt, ct0, acc);
    const bf16_t* hb = hband + (size_t)k * LL * 512;
#pragma unroll
    for (int t = 0; t < 8; ++t) {
      int col = (ct0 + t) * 16 + (lane & 15);
      float bv = b_mod[k * 512 + col];
#pragma unroll
      for (int r = 0; r < 8; ++r) {
        int row = rt * 16 + r + kh8;
        int gRow = gRow0 + row;
        int l = gRow & (LL - 1);
        float m = fmaxf(acc[t][r] + bv + (float)hb[(size_t)l * 512 + col], 0.f);
        if (k == 0) {
          s_hv[(size_t)row * 512 + col] = f2bf(m);
#pragma unroll
          for (int o = 0; o < 3; ++o) tot[r][o] += m * w_out[col * 3 + o];
        } else {
          float s = m + (float)s_hv[(size_t)row * 512 + col];
          s_hv[(size_t)row * 512 + col] = f2bf(s);
        }
      }
    }
    __syncthreads();
    if (k > 0) {
      // hv = relu((m + hv) @ w_hv[k-1] + b_hv[k-1])
      gemm512(s_hv, whv_p + (size_t)(k - 1) * 512 * 512, rt, ct0, acc);
      __syncthreads();  // all reads of s before overwrite
#pragma unroll
      for (int t = 0; t < 8; ++t) {
        int col = (ct0 + t) * 16 + (lane & 15);
        float bv = b_hv[(k - 1) * 512 + col];
        const float* wo = w_out + (size_t)k * 512 * 3;
#pragma unroll
        for (int r = 0; r < 8; ++r) {
          int row = rt * 16 + r + kh8;
          float hvn = fmaxf(acc[t][r] + bv, 0.f);
          s_hv[(size_t)row * 512 + col] = f2bf(hvn);
#pragma unroll
          for (int o = 0; o < 3; ++o) tot[r][o] += hvn * wo[col * 3 + o];
        }
      }
      __syncthreads();
    }
  }

  // reduce tot across the 16-lane half-groups, then atomicAdd to d_out
#pragma unroll
  for (int off = 1; off < 16; off <<= 1)
#pragma unroll
    for (int r = 0; r < 8; ++r)
#pragma unroll
      for (int o = 0; o < 3; ++o) tot[r][o] += __shfl_xor(tot[r][o], off, 32);
  if ((lane & 15) == 0) {
#pragma unroll
    for (int r = 0; r < 8; ++r) {
      int gRow = gRow0 + rt * 16 + r + kh8;
#pragma unroll
      for (int o = 0; o < 3; ++o)
        atomicAdd(&outp[(size_t)gRow * 3 + o], tot[r][o]);
    }
  }
}

// ---------------------------------------------------------------------------
extern "C" void kernel_launch(void* const* d_in, const int* in_sizes, int n_in,
                              void* d_out, int out_size, void* d_ws, size_t ws_size,
                              hipStream_t stream) {
  const float* x        = (const float*)d_in[0];
  const float* tokens   = (const float*)d_in[1];
  const float* w_query  = (const float*)d_in[2];
  const float* b_query  = (const float*)d_in[3];
  const float* w_to_q   = (const float*)d_in[4];
  const float* w_to_kv  = (const float*)d_in[5];
  const float* w_to_out = (const float*)d_in[6];
  const float* b_to_out = (const float*)d_in[7];
  const float* w_band   = (const float*)d_in[8];
  const float* b_band   = (const float*)d_in[9];
  const float* w_mod    = (const float*)d_in[10];
  const float* b_mod    = (const float*)d_in[11];
  const float* w_hv     = (const float*)d_in[12];
  const float* b_hv     = (const float*)d_in[13];
  const float* w_out    = (const float*)d_in[14];
  const float* b_out    = (const float*)d_in[15];
  const int*   Dd       = (const int*)d_in[16];
  const int*   Hh       = (const int*)d_in[17];
  const int*   Ww       = (const int*)d_in[18];
  const int*   Tt       = (const int*)d_in[19];
  float* outp = (float*)d_out;

  // workspace carve-up (bytes, 256B aligned)
  uint8_t* ws = (uint8_t*)d_ws;
  size_t off = 0;
  auto carve = [&](size_t bytes) {
    size_t o = off;
    off = (off + bytes + 255) & ~(size_t)255;
    return o;
  };
  bf16_t* wq_p     = (bf16_t*)(ws + carve((size_t)128 * 512 * 2));
  bf16_t* wband_p  = (bf16_t*)(ws + carve((size_t)3 * 128 * 512 * 2));
  bf16_t* wtoq_p   = (bf16_t*)(ws + carve((size_t)512 * 128 * 2));
  bf16_t* wtokv_p  = (bf16_t*)(ws + carve((size_t)512 * 256 * 2));
  bf16_t* wtoout_p = (bf16_t*)(ws + carve((size_t)128 * 512 * 2));
  bf16_t* wmod_p   = (bf16_t*)(ws + carve((size_t)3 * 512 * 512 * 2));
  bf16_t* whv_p    = (bf16_t*)(ws + carve((size_t)2 * 512 * 512 * 2));
  bf16_t* tok_bf   = (bf16_t*)(ws + carve((size_t)BB * MM * HIDD * 2));
  bf16_t* qb       = (bf16_t*)(ws + carve((size_t)LL * 128 * 2));
  bf16_t* kb       = (bf16_t*)(ws + carve((size_t)BB * NHEAD * MM * DHH * 2));
  bf16_t* vT       = (bf16_t*)(ws + carve((size_t)BB * NHEAD * DHH * MM * 2));
  bf16_t* hband    = (bf16_t*)(ws + carve((size_t)3 * LL * 512 * 2));
  bf16_t* attn_out = (bf16_t*)(ws + carve((size_t)BB * LL * 128 * 2));
  float*  tpos     = (float*)(ws + carve((size_t)LL * 4));

  auto packW = [&](const float* src, bf16_t* dst, int K, int N) {
    int total = (K / 32) * (N / 16) * 32;
    pack_weight_kernel<<<(total + 255) / 256, 256, 0, stream>>>(src, dst, K, N);
  };
  packW(w_query, wq_p, 128, 512);
  for (int k = 0; k < 3; ++k)
    packW(w_band + (size_t)k * 128 * 512, wband_p + (size_t)k * 128 * 512, 128, 512);
  packW(w_to_q, wtoq_p, 512, 128);
  packW(w_to_kv, wtokv_p, 512, 256);
  packW(w_to_out, wtoout_p, 128, 512);
  for (int k = 0; k < 3; ++k)
    packW(w_mod + (size_t)k * 512 * 512, wmod_p + (size_t)k * 512 * 512, 512, 512);
  for (int k = 0; k < 2; ++k)
    packW(w_hv + (size_t)k * 512 * 512, whv_p + (size_t)k * 512 * 512, 512, 512);

  {
    int n = BB * MM * HIDD;
    f32_to_bf16_kernel<<<(n + 255) / 256, 256, 0, stream>>>(tokens, tok_bf, n);
  }
  tpos_kernel<<<LL / 256, 256, 0, stream>>>(x, tpos, Dd, Hh, Ww, Tt);

  feature_kernel<<<LL / 32, 256, 0, stream>>>(x, wq_p, b_query, wband_p, b_band,
                                              wtoq_p, qb, hband);
  kv_kernel<<<(BB * MM) / 32, 256, 0, stream>>>(tok_bf, wtokv_p, kb, vT);

  attn_kernel<<<dim3(LL / 64, BB * NHEAD), 128, 0, stream>>>(qb, kb, vT, tpos, attn_out);

  {
    int n = BB * LL * ODIMM;
    out_init_kernel<<<(n + 255) / 256, 256, 0, stream>>>(outp, b_out);
  }
  chain_kernel<<<(BB * LL) / 32, 256, 0, stream>>>(attn_out, wtoout_p, b_to_out, hband,
                                                   wmod_p, b_mod, whv_p, b_hv, w_out, outp);

  (void)in_sizes; (void)n_in; (void)out_size; (void)ws_size;
}
